// LDALoss_55800215110096
// MI455X (gfx1250) — matrix operand order repaired and checked by reading, not compiled, not run on hardware
//
#include <hip/hip_runtime.h>
#include <hip/hip_bf16.h>

typedef __attribute__((ext_vector_type(4)))  float   v4f;
typedef __attribute__((ext_vector_type(8)))  float   v8f;
typedef __attribute__((ext_vector_type(4)))  int     v4i;
typedef __attribute__((ext_vector_type(8)))  __bf16  v8bf;
typedef __attribute__((ext_vector_type(16))) __bf16  v16bf;

#define BATCH 4096
#define NCLS  10000
#define KDIM  1024
#define BM 128
#define BN 128
#define BK 32
#define LDSTR 40           // padded LDS row stride (bf16 elems): 80B rows, conflict-free, 16B aligned
#define TILE_E (BM * LDSTR)
#define NK (KDIM / BK)     // 32 K-chunks

union FragBF { v16bf v; v8bf h[2]; };

// ---- gfx1250 async-copy plumbing (guarded: falls back to sync LDS copy) ----
#if defined(__HIP_DEVICE_COMPILE__) && __has_builtin(__builtin_amdgcn_global_load_async_to_lds_b128)
#define HAVE_ASYNC 1
#else
#define HAVE_ASYNC 0
#endif

#if HAVE_ASYNC
typedef __attribute__((address_space(1))) v4i g_v4i;   // global-AS int4
typedef __attribute__((address_space(3))) v4i l_v4i;   // LDS-AS int4
# if __has_builtin(__builtin_amdgcn_s_wait_asynccnt)
#  define WAIT_ASYNC(N) __builtin_amdgcn_s_wait_asynccnt(N)
# else
#  define WAIT_ASYNC(N) asm volatile("s_wait_asynccnt " #N ::: "memory")
# endif
#else
# define WAIT_ASYNC(N) do { } while (0)
#endif

__device__ __forceinline__ void async_cp16(const __bf16* g, __bf16* l) {
#if HAVE_ASYNC
  __builtin_amdgcn_global_load_async_to_lds_b128((g_v4i*)g, (l_v4i*)l, 0, 0);
#else
  *(v8bf*)l = *(const v8bf*)g;
#endif
}

// ---- out[row] = 0.5 * sum(x[row,:]^2), one wave per row ----
__global__ __launch_bounds__(256) void rowsq_half_kernel(const float* __restrict__ x,
                                                         float* __restrict__ out, int nrows) {
  int gwave = (int)((blockIdx.x * blockDim.x + threadIdx.x) >> 5);
  int lane  = threadIdx.x & 31;
  if (gwave >= nrows) return;
  const v4f* p = (const v4f*)(x + (size_t)gwave * KDIM);
  float s = 0.f;
#pragma unroll
  for (int i = 0; i < 8; ++i) {
    v4f v = p[i * 32 + lane];
    s += v[0]*v[0] + v[1]*v[1] + v[2]*v[2] + v[3]*v[3];
  }
#pragma unroll
  for (int off = 16; off > 0; off >>= 1) s += __shfl_xor(s, off, 32);
  if (lane == 0) out[gwave] = 0.5f * s;
}

// ---- one-pass fp32 -> (hi,lo) bf16 split; each element converted exactly once ----
__global__ __launch_bounds__(256) void split_bf16_kernel(const float* __restrict__ x,
                                                         __bf16* __restrict__ hi,
                                                         __bf16* __restrict__ lo,
                                                         long n) {
  long i = ((long)blockIdx.x * 256 + threadIdx.x) * 8;
  if (i >= n) return;
  v4f a = *(const v4f*)(x + i);
  v4f b = *(const v4f*)(x + i + 4);
  v8bf vh, vl;
#pragma unroll
  for (int j = 0; j < 4; ++j) {
    float t = a[j]; __bf16 h = (__bf16)t; vh[j] = h; vl[j] = (__bf16)(t - (float)h);
  }
#pragma unroll
  for (int j = 0; j < 4; ++j) {
    float t = b[j]; __bf16 h = (__bf16)t; vh[4+j] = h; vl[4+j] = (__bf16)(t - (float)h);
  }
  *(v8bf*)(hi + i) = vh;
  *(v8bf*)(lo + i) = vl;
}

// ---- main GEMM: async global->LDS double-buffered, pure WMMA inner loop ----
__global__ __launch_bounds__(256) void lda_gemm_wmma_async(
    const __bf16* __restrict__ Ahg, const __bf16* __restrict__ Alg,
    const __bf16* __restrict__ Bhg, const __bf16* __restrict__ Blg,
    const float* __restrict__ fsh, const float* __restrict__ csh,
    float* __restrict__ out) {
  __shared__ __bf16 smem[2 * 4 * TILE_E];   // [buf][Ah,Al,Bh,Bl] = 80 KB

  const int tid  = threadIdx.x;
  const int row0 = blockIdx.y * BM;
  const int col0 = blockIdx.x * BN;

  // copy mapping: thread t moves 16B segment (t&3) of rows (t>>2) and (t>>2)+64
  const int prow = tid >> 2;                 // 0..63
  const int pel  = (tid & 3) * 8;            // element offset of 16B segment
  const int ar0 = row0 + prow, ar1 = ar0 + 64;
  int br0 = col0 + prow;      br0 = br0 < NCLS ? br0 : (NCLS - 1);
  int br1 = col0 + prow + 64; br1 = br1 < NCLS ? br1 : (NCLS - 1);
  const int lofs = prow * LDSTR + pel;

  const int wave = tid >> 5;
  const int wm   = wave >> 2;
  const int wn   = wave & 3;
  const int lane = tid & 31;
  const int half = lane >> 4;
  const int l15  = lane & 15;

  v8f acc[4][2];
#pragma unroll
  for (int mt = 0; mt < 4; ++mt)
#pragma unroll
    for (int nt = 0; nt < 2; ++nt)
#pragma unroll
      for (int r = 0; r < 8; ++r) acc[mt][nt][r] = 0.f;

#define ISSUE_CHUNK(KC, BUF)                                                              \
  do {                                                                                    \
    __bf16* lb = smem + (BUF) * (4 * TILE_E);                                             \
    const int ke = (KC) * BK + pel;                                                       \
    async_cp16(Ahg + (size_t)ar0 * KDIM + ke, lb + 0 * TILE_E + lofs);                    \
    async_cp16(Ahg + (size_t)ar1 * KDIM + ke, lb + 0 * TILE_E + lofs + 64 * LDSTR);       \
    async_cp16(Alg + (size_t)ar0 * KDIM + ke, lb + 1 * TILE_E + lofs);                    \
    async_cp16(Alg + (size_t)ar1 * KDIM + ke, lb + 1 * TILE_E + lofs + 64 * LDSTR);       \
    async_cp16(Bhg + (size_t)br0 * KDIM + ke, lb + 2 * TILE_E + lofs);                    \
    async_cp16(Bhg + (size_t)br1 * KDIM + ke, lb + 2 * TILE_E + lofs + 64 * LDSTR);       \
    async_cp16(Blg + (size_t)br0 * KDIM + ke, lb + 3 * TILE_E + lofs);                    \
    async_cp16(Blg + (size_t)br1 * KDIM + ke, lb + 3 * TILE_E + lofs + 64 * LDSTR);       \
  } while (0)

  ISSUE_CHUNK(0, 0);

  for (int kc = 0; kc < NK; ++kc) {
    if (kc + 1 < NK) {
      ISSUE_CHUNK(kc + 1, (kc + 1) & 1);
      WAIT_ASYNC(8);          // retire the 8 older copies (chunk kc); async loads done in order
    } else {
      WAIT_ASYNC(0);
    }
    __syncthreads();

    const __bf16* lb  = smem + (kc & 1) * (4 * TILE_E);
    const __bf16* lAh = lb + 0 * TILE_E;
    const __bf16* lAl = lb + 1 * TILE_E;
    const __bf16* lBh = lb + 2 * TILE_E;
    const __bf16* lBl = lb + 3 * TILE_E;

#pragma unroll
    for (int nt = 0; nt < 2; ++nt) {
      const int bn = wn * 32 + nt * 16;
      FragBF Bh, Bl;
      const __bf16* pb  = &lBh[(bn + l15) * LDSTR + half * 8];
      const __bf16* pbl = &lBl[(bn + l15) * LDSTR + half * 8];
      Bh.h[0] = *(const v8bf*)pb;   Bh.h[1] = *(const v8bf*)(pb + 16);
      Bl.h[0] = *(const v8bf*)pbl;  Bl.h[1] = *(const v8bf*)(pbl + 16);
#pragma unroll
      for (int mt = 0; mt < 4; ++mt) {
        const int bm = wm * 64 + mt * 16;
        FragBF Ah, Al;
        const __bf16* pa  = &lAh[(bm + l15) * LDSTR + half * 8];
        const __bf16* pal = &lAl[(bm + l15) * LDSTR + half * 8];
        Ah.h[0] = *(const v8bf*)pa;   Ah.h[1] = *(const v8bf*)(pa + 16);
        Al.h[0] = *(const v8bf*)pal;  Al.h[1] = *(const v8bf*)(pal + 16);

        acc[mt][nt] = __builtin_amdgcn_wmma_f32_16x16x32_bf16(
            false, Ah.v, false, Bh.v, (short)0, acc[mt][nt], false, false);
        acc[mt][nt] = __builtin_amdgcn_wmma_f32_16x16x32_bf16(
            false, Ah.v, false, Bl.v, (short)0, acc[mt][nt], false, false);
        acc[mt][nt] = __builtin_amdgcn_wmma_f32_16x16x32_bf16(
            false, Al.v, false, Bh.v, (short)0, acc[mt][nt], false, false);
      }
    }
    __syncthreads();
  }

#pragma unroll
  for (int nt = 0; nt < 2; ++nt) {
    const int gn = col0 + wn * 32 + nt * 16 + l15;
    if (gn < NCLS) {
      const float c = csh[gn];
#pragma unroll
      for (int mt = 0; mt < 4; ++mt) {
#pragma unroll
        for (int r = 0; r < 8; ++r) {
          const int gm = row0 + wm * 64 + mt * 16 + half * 8 + r;
          out[(size_t)gm * NCLS + gn] = acc[mt][nt][r] - fsh[gm] - c;
        }
      }
    }
  }
#undef ISSUE_CHUNK
}

// ---- fallback GEMM (round-1 style, converts in-kernel; used only if ws too small) ----
__global__ __launch_bounds__(256) void lda_gemm_wmma_inline(const float* __restrict__ feat,
                                                            const float* __restrict__ cent,
                                                            const float* __restrict__ fsh,
                                                            const float* __restrict__ csh,
                                                            float* __restrict__ out) {
  __shared__ __bf16 lAh[TILE_E];
  __shared__ __bf16 lAl[TILE_E];
  __shared__ __bf16 lBh[TILE_E];
  __shared__ __bf16 lBl[TILE_E];

  const int tid  = threadIdx.x;
  const int row0 = blockIdx.y * BM;
  const int col0 = blockIdx.x * BN;
  const int lrow = tid >> 1;
  const int lcol = (tid & 1) << 4;
  const float* gA = feat + (size_t)(row0 + lrow) * KDIM + lcol;
  const int crow_raw = col0 + lrow;
  const int crow = crow_raw < NCLS ? crow_raw : (NCLS - 1);
  const float* gB = cent + (size_t)crow * KDIM + lcol;

  v4f ra[4], rb[4];
#pragma unroll
  for (int i = 0; i < 4; ++i) { ra[i] = *(const v4f*)(gA + i*4); rb[i] = *(const v4f*)(gB + i*4); }

  const int wave = tid >> 5, wm = wave >> 2, wn = wave & 3;
  const int lane = tid & 31, half = lane >> 4, l15 = lane & 15;

  v8f acc[4][2];
#pragma unroll
  for (int mt = 0; mt < 4; ++mt)
#pragma unroll
    for (int nt = 0; nt < 2; ++nt)
#pragma unroll
      for (int r = 0; r < 8; ++r) acc[mt][nt][r] = 0.f;

  for (int kc = 0; kc < NK; ++kc) {
#pragma unroll
    for (int g = 0; g < 2; ++g) {
      v8bf vh, vl;
#pragma unroll
      for (int i = 0; i < 8; ++i) {
        float x = ra[g*2 + (i>>2)][i&3]; __bf16 h = (__bf16)x;
        vh[i] = h; vl[i] = (__bf16)(x - (float)h);
      }
      *(v8bf*)&lAh[lrow*LDSTR + lcol + g*8] = vh;
      *(v8bf*)&lAl[lrow*LDSTR + lcol + g*8] = vl;
#pragma unroll
      for (int i = 0; i < 8; ++i) {
        float x = rb[g*2 + (i>>2)][i&3]; __bf16 h = (__bf16)x;
        vh[i] = h; vl[i] = (__bf16)(x - (float)h);
      }
      *(v8bf*)&lBh[lrow*LDSTR + lcol + g*8] = vh;
      *(v8bf*)&lBl[lrow*LDSTR + lcol + g*8] = vl;
    }
    __syncthreads();

    if (kc + 1 < NK) {
#pragma unroll
      for (int i = 0; i < 4; ++i) {
        ra[i] = *(const v4f*)(gA + (kc+1)*BK + i*4);
        rb[i] = *(const v4f*)(gB + (kc+1)*BK + i*4);
      }
    }

#pragma unroll
    for (int nt = 0; nt < 2; ++nt) {
      const int bn = wn * 32 + nt * 16;
      FragBF Bh, Bl;
      const __bf16* pb  = &lBh[(bn + l15)*LDSTR + half*8];
      const __bf16* pbl = &lBl[(bn + l15)*LDSTR + half*8];
      Bh.h[0] = *(const v8bf*)pb;   Bh.h[1] = *(const v8bf*)(pb + 16);
      Bl.h[0] = *(const v8bf*)pbl;  Bl.h[1] = *(const v8bf*)(pbl + 16);
#pragma unroll
      for (int mt = 0; mt < 4; ++mt) {
        const int bm = wm * 64 + mt * 16;
        FragBF Ah, Al;
        const __bf16* pa  = &lAh[(bm + l15)*LDSTR + half*8];
        const __bf16* pal = &lAl[(bm + l15)*LDSTR + half*8];
        Ah.h[0] = *(const v8bf*)pa;   Ah.h[1] = *(const v8bf*)(pa + 16);
        Al.h[0] = *(const v8bf*)pal;  Al.h[1] = *(const v8bf*)(pal + 16);
        acc[mt][nt] = __builtin_amdgcn_wmma_f32_16x16x32_bf16(false, Ah.v, false, Bh.v, (short)0, acc[mt][nt], false, false);
        acc[mt][nt] = __builtin_amdgcn_wmma_f32_16x16x32_bf16(false, Ah.v, false, Bl.v, (short)0, acc[mt][nt], false, false);
        acc[mt][nt] = __builtin_amdgcn_wmma_f32_16x16x32_bf16(false, Al.v, false, Bh.v, (short)0, acc[mt][nt], false, false);
      }
    }
    __syncthreads();
  }

#pragma unroll
  for (int nt = 0; nt < 2; ++nt) {
    const int gn = col0 + wn * 32 + nt * 16 + l15;
    if (gn < NCLS) {
      const float c = csh[gn];
#pragma unroll
      for (int mt = 0; mt < 4; ++mt)
#pragma unroll
        for (int r = 0; r < 8; ++r) {
          const int gm = row0 + wm * 64 + mt * 16 + half * 8 + r;
          out[(size_t)gm * NCLS + gn] = acc[mt][nt][r] - fsh[gm] - c;
        }
    }
  }
}

extern "C" void kernel_launch(void* const* d_in, const int* in_sizes, int n_in,
                              void* d_out, int out_size, void* d_ws, size_t ws_size,
                              hipStream_t stream) {
  const float* feat = (const float*)d_in[0];   // [4096, 1024] f32
  const float* cent = (const float*)d_in[1];   // [10000, 1024] f32
  float* out = (float*)d_out;

  char* ws = (char*)d_ws;
  float* fsh = (float*)ws;                     // 4096 floats
  float* csh = fsh + BATCH;                    // 10000 floats

  rowsq_half_kernel<<<BATCH / 8, 256, 0, stream>>>(feat, fsh, BATCH);
  rowsq_half_kernel<<<(NCLS + 7) / 8, 256, 0, stream>>>(cent, csh, NCLS);

  const size_t FEAT_E = (size_t)BATCH * KDIM;          // 4,194,304
  const size_t CENT_E = (size_t)NCLS * KDIM;           // 10,240,000
  const size_t SPLIT_OFF = 57344;                      // past fsh/csh, 4KB aligned
  const size_t NEED = SPLIT_OFF + 2 * (FEAT_E + CENT_E) * sizeof(__bf16);

  dim3 grid((NCLS + BN - 1) / BN, BATCH / BM);         // (79, 32)

  if (ws_size >= NEED) {
    __bf16* fh = (__bf16*)(ws + SPLIT_OFF);
    __bf16* fl = fh + FEAT_E;
    __bf16* ch = fl + FEAT_E;
    __bf16* cl = ch + CENT_E;
    split_bf16_kernel<<<(int)(FEAT_E / 8 / 256), 256, 0, stream>>>(feat, fh, fl, (long)FEAT_E);
    split_bf16_kernel<<<(int)(CENT_E / 8 / 256), 256, 0, stream>>>(cent, ch, cl, (long)CENT_E);
    lda_gemm_wmma_async<<<grid, 256, 0, stream>>>(fh, fl, ch, cl, fsh, csh, out);
  } else {
    lda_gemm_wmma_inline<<<grid, 256, 0, stream>>>(feat, cent, fsh, csh, out);
  }
}
